// MoE_13993003450834
// MI455X (gfx1250) — compile-verified, matrix operand order checked by reading, not run on hardware
//
#include <hip/hip_runtime.h>
#include <hip/hip_bf16.h>

// ---------------------------------------------------------------------------
// MoE top-2, fused per-expert MLP with bf16 WMMA (gfx1250 / MI455X)
// B=4,S=2048 -> T=8192 tokens, H=512, I=1024, E=8, TOP_K=2
// TM=32 rows per block: 2 M-subtiles share every B fragment (2x arithmetic
// intensity vs TM=16 -> ~1 GB L2 weight traffic instead of 2 GB).
// ---------------------------------------------------------------------------

typedef __attribute__((ext_vector_type(16))) __bf16        v16bf;
typedef __attribute__((ext_vector_type(8)))  float         v8f;
typedef __attribute__((ext_vector_type(4)))  unsigned int  u32x4;

union Frag16 { v16bf v; u32x4 q[2]; };

#define MOE_T 8192
#define MOE_H 512
#define MOE_I 1024
#define MOE_E 8
#define TM    32                      // gathered rows per block (2 x 16)

__device__ __forceinline__ unsigned short f2bf(float f) {
    unsigned int u = __builtin_bit_cast(unsigned int, f);
    u += 0x7FFFu + ((u >> 16) & 1u);           // round-to-nearest-even
    return (unsigned short)(u >> 16);
}

// -------------------------- prep: x -> bf16 --------------------------------
__global__ __launch_bounds__(256) void moe_prep_x(const float* __restrict__ x,
                                                  unsigned short* __restrict__ xbf) {
    int i = blockIdx.x * 256 + threadIdx.x;     // grid covers T*H exactly
    xbf[i] = f2bf(x[i]);
}

// ------------------- prep: w1 [E][H][I] -> W1^T bf16 [E][I][H] -------------
__global__ __launch_bounds__(256) void moe_prep_w1t(const float* __restrict__ w1,
                                                    unsigned short* __restrict__ w1t) {
    size_t idx = (size_t)blockIdx.x * 256 + threadIdx.x;   // E*I*H
    int e   = (int)(idx >> 19);
    int rem = (int)(idx & ((1u << 19) - 1));
    int i   = rem >> 9;            // 0..1023
    int h   = rem & 511;           // 0..511
    w1t[idx] = f2bf(w1[((size_t)e << 19) + (size_t)h * MOE_I + i]);
}

// ------------------- prep: w2 [E][I][H] -> W2^T bf16 [E][H][I] -------------
__global__ __launch_bounds__(256) void moe_prep_w2t(const float* __restrict__ w2,
                                                    unsigned short* __restrict__ w2t) {
    size_t idx = (size_t)blockIdx.x * 256 + threadIdx.x;   // E*H*I
    int e   = (int)(idx >> 19);
    int rem = (int)(idx & ((1u << 19) - 1));
    int h   = rem >> 10;           // 0..511
    int i   = rem & 1023;          // 0..1023
    w2t[idx] = f2bf(w2[((size_t)e << 19) + (size_t)i * MOE_H + h]);
}

// ------------------------------- router ------------------------------------
__global__ __launch_bounds__(256) void moe_router(const float* __restrict__ x,
                                                  const float* __restrict__ gw,
                                                  float* __restrict__ logits_out,
                                                  int*   __restrict__ slotExp,
                                                  float* __restrict__ slotW,
                                                  int*   __restrict__ counts) {
    __shared__ float gs[MOE_E][MOE_H];
    int tid = threadIdx.x;
    for (int i = tid; i < MOE_E * MOE_H; i += 256) gs[i >> 9][i & 511] = gw[i];
    __syncthreads();

    int t = blockIdx.x * 256 + tid;
    const float* xr = x + (size_t)t * MOE_H;
    float acc[MOE_E];
#pragma unroll
    for (int e = 0; e < MOE_E; ++e) acc[e] = 0.f;

    for (int h = 0; h < MOE_H; h += 4) {
        float4 xv = *(const float4*)(xr + h);   // uniform h -> LDS broadcast reads
#pragma unroll
        for (int e = 0; e < MOE_E; ++e)
            acc[e] += xv.x * gs[e][h] + xv.y * gs[e][h + 1] +
                      xv.z * gs[e][h + 2] + xv.w * gs[e][h + 3];
    }
#pragma unroll
    for (int e = 0; e < MOE_E; ++e) logits_out[(size_t)t * MOE_E + e] = acc[e];

    // softmax (fp32)
    float mx = acc[0];
#pragma unroll
    for (int e = 1; e < MOE_E; ++e) mx = fmaxf(mx, acc[e]);
    float p[MOE_E], s = 0.f;
#pragma unroll
    for (int e = 0; e < MOE_E; ++e) { p[e] = __expf(acc[e] - mx); s += p[e]; }
    float inv = 1.f / s;
#pragma unroll
    for (int e = 0; e < MOE_E; ++e) p[e] *= inv;

    // top-2 (ties -> lowest index, strict >)
    int i0 = 0; float p0 = p[0];
#pragma unroll
    for (int e = 1; e < MOE_E; ++e) if (p[e] > p0) { p0 = p[e]; i0 = e; }
    int i1 = (i0 == 0) ? 1 : 0; float p1 = p[i1];
#pragma unroll
    for (int e = 0; e < MOE_E; ++e)
        if (e != i0 && p[e] > p1) { p1 = p[e]; i1 = e; }

    float norm = 0.5f / (p0 + p1 + 1e-20f);     // /sum, /TOP_K
    slotExp[t * 2 + 0] = i0;  slotW[t * 2 + 0] = p0 * norm;
    slotExp[t * 2 + 1] = i1;  slotW[t * 2 + 1] = p1 * norm;
    atomicAdd(&counts[i0], 1);
    atomicAdd(&counts[i1], 1);
}

// ----------------------- scan counts -> offsets ----------------------------
__global__ void moe_scan(const int* __restrict__ counts,
                         int* __restrict__ offsets, int* __restrict__ cursors) {
    if (threadIdx.x == 0) {
        int s = 0;
        for (int e = 0; e < MOE_E; ++e) { offsets[e] = s; s += counts[e]; cursors[e] = 0; }
    }
}

// --------------------- build gathered row lists ----------------------------
__global__ __launch_bounds__(256) void moe_build(const int* __restrict__ slotExp,
                                                 const float* __restrict__ slotW,
                                                 const int* __restrict__ offsets,
                                                 int* __restrict__ cursors,
                                                 int* __restrict__ rowTok,
                                                 float* __restrict__ rowWgt) {
    int t = blockIdx.x * 256 + threadIdx.x;
    if (t >= MOE_T) return;
#pragma unroll
    for (int s = 0; s < 2; ++s) {
        int e = slotExp[t * 2 + s];
        int pos = atomicAdd(&cursors[e], 1);
        int idx = offsets[e] + pos;
        rowTok[idx] = t;
        rowWgt[idx] = slotW[t * 2 + s];
    }
}

// ---------------- fused expert MLP over a 32-row gathered tile -------------
//   GEMM1: A[32x512] x W1[512x1024] -> relu -> h (bf16, LDS)
//   GEMM2: h[32x1024] x W2[1024x512] -> weight-scaled atomic add to out
__global__ __launch_bounds__(256) void moe_expert(
    const unsigned short* __restrict__ xbf,   // [T][512] bf16
    const unsigned short* __restrict__ w1t,   // [E][1024][512] bf16 (W1^T)
    const unsigned short* __restrict__ w2t,   // [E][512][1024] bf16 (W2^T)
    const float* __restrict__ b1,             // [E][1024]
    const float* __restrict__ b2,             // [E][512]
    const int*   __restrict__ rowTok,         // [2T]
    const float* __restrict__ rowWgt,         // [2T]
    const int*   __restrict__ counts,         // [E]
    const int*   __restrict__ offsets,        // [E]
    float* __restrict__ out)                  // [T][512]
{
    const int e   = blockIdx.y;
    const int cnt = counts[e];
    const int r0  = blockIdx.x * TM;
    if (r0 >= cnt) return;
    const int off = offsets[e];

    __shared__ unsigned short Ash[TM][MOE_H + 8];   // +8 bf16 pad -> conflict-free b128
    __shared__ unsigned short Hsh[TM][MOE_I + 8];
    __shared__ int   tokS[TM];
    __shared__ float wS[TM];

    const int tid  = threadIdx.x;
    const int wave = tid >> 5;
    const int lane = tid & 31;
    const int l15  = lane & 15;
    const int hi8  = (lane >> 4) << 3;   // 0 or 8 (C/D row offset; also A/B K-base)

    if (tid < TM) {
        int r = r0 + tid;
        if (r < cnt) { tokS[tid] = rowTok[off + r]; wS[tid] = rowWgt[off + r]; }
        else         { tokS[tid] = -1;              wS[tid] = 0.f; }
    }
    __syncthreads();

    // stage gathered A rows (TM x 512 bf16) into LDS, 16B per transfer
    for (int idx = tid; idx < TM * (MOE_H / 8); idx += 256) {
        int r = idx >> 6;
        int c = (idx & 63) * 8;
        int t = tokS[r];
        u32x4 val = (u32x4)(0u);
        if (t >= 0) val = *(const u32x4*)(xbf + (size_t)t * MOE_H + c);
        *(u32x4*)(&Ash[r][c]) = val;
    }
    __syncthreads();

    // ---------------- GEMM1: 64 n-tiles of 16, K = 512 ----------------
    const unsigned short* W1e = w1t + (size_t)e * MOE_I * MOE_H;
    for (int nt = wave; nt < MOE_I / 16; nt += 8) {
        const int n0 = nt * 16;
        const float b1v = b1[(size_t)e * MOE_I + n0 + l15];
        v8f acc0 = (v8f)(0.0f);
        v8f acc1 = (v8f)(0.0f);
        for (int kt = 0; kt < MOE_H / 32; ++kt) {
            const int k0 = kt * 32 + hi8;
            Frag16 bF, a0, a1;
            const unsigned short* bp = W1e + (size_t)(n0 + l15) * MOE_H + k0;
            bF.q[0] = *(const u32x4*)(bp);
            bF.q[1] = *(const u32x4*)(bp + 16);
            const unsigned short* ap0 = &Ash[l15][k0];
            a0.q[0] = *(const u32x4*)(ap0);
            a0.q[1] = *(const u32x4*)(ap0 + 16);
            const unsigned short* ap1 = &Ash[16 + l15][k0];
            a1.q[0] = *(const u32x4*)(ap1);
            a1.q[1] = *(const u32x4*)(ap1 + 16);
            acc0 = __builtin_amdgcn_wmma_f32_16x16x32_bf16(
                       false, a0.v, false, bF.v, (short)0, acc0, false, false);
            acc1 = __builtin_amdgcn_wmma_f32_16x16x32_bf16(
                       false, a1.v, false, bF.v, (short)0, acc1, false, false);
        }
#pragma unroll
        for (int g = 0; g < 8; ++g) {
            float v0 = acc0[g] + b1v;
            float v1 = acc1[g] + b1v;
            v0 = v0 > 0.f ? v0 : 0.f;               // ReLU
            v1 = v1 > 0.f ? v1 : 0.f;
            Hsh[g + hi8][n0 + l15]      = f2bf(v0); // C layout: row g(+8), col l15
            Hsh[16 + g + hi8][n0 + l15] = f2bf(v1);
        }
    }
    __syncthreads();

    // ---------------- GEMM2: 32 n-tiles of 16, K = 1024 ----------------
    const unsigned short* W2e = w2t + (size_t)e * MOE_H * MOE_I;
    for (int nt = wave; nt < MOE_H / 16; nt += 8) {
        const int n0 = nt * 16;
        const float b2v = b2[(size_t)e * MOE_H + n0 + l15];
        v8f acc0 = (v8f)(0.0f);
        v8f acc1 = (v8f)(0.0f);
        for (int kt = 0; kt < MOE_I / 32; ++kt) {
            const int k0 = kt * 32 + hi8;
            Frag16 bF, a0, a1;
            const unsigned short* bp = W2e + (size_t)(n0 + l15) * MOE_I + k0;
            bF.q[0] = *(const u32x4*)(bp);
            bF.q[1] = *(const u32x4*)(bp + 16);
            const unsigned short* ap0 = &Hsh[l15][k0];
            a0.q[0] = *(const u32x4*)(ap0);
            a0.q[1] = *(const u32x4*)(ap0 + 16);
            const unsigned short* ap1 = &Hsh[16 + l15][k0];
            a1.q[0] = *(const u32x4*)(ap1);
            a1.q[1] = *(const u32x4*)(ap1 + 16);
            acc0 = __builtin_amdgcn_wmma_f32_16x16x32_bf16(
                       false, a0.v, false, bF.v, (short)0, acc0, false, false);
            acc1 = __builtin_amdgcn_wmma_f32_16x16x32_bf16(
                       false, a1.v, false, bF.v, (short)0, acc1, false, false);
        }
#pragma unroll
        for (int g = 0; g < 8; ++g) {
            int ra = g + hi8;
            int rb = 16 + g + hi8;
            int ta = tokS[ra];
            int tb = tokS[rb];
            if (ta >= 0)
                atomicAdd(out + (size_t)ta * MOE_H + n0 + l15,
                          (acc0[g] + b2v) * wS[ra]);       // exactly 2 adds/elem
            if (tb >= 0)
                atomicAdd(out + (size_t)tb * MOE_H + n0 + l15,
                          (acc1[g] + b2v) * wS[rb]);
        }
    }
}

// ---------------------------------------------------------------------------
extern "C" void kernel_launch(void* const* d_in, const int* in_sizes, int n_in,
                              void* d_out, int out_size, void* d_ws, size_t ws_size,
                              hipStream_t stream) {
    const float* x  = (const float*)d_in[0];   // [T,512]
    const float* gw = (const float*)d_in[1];   // [8,512]
    const float* w1 = (const float*)d_in[2];   // [8,512,1024]
    const float* b1 = (const float*)d_in[3];   // [8,1024]
    const float* w2 = (const float*)d_in[4];   // [8,1024,512]
    const float* b2 = (const float*)d_in[5];   // [8,512]

    float* out    = (float*)d_out;                        // [T,512]
    float* logits = (float*)d_out + (size_t)MOE_T * MOE_H;// [T,8]

    char* ws = (char*)d_ws;
    const size_t SZ_XBF = (size_t)MOE_T * MOE_H * 2;              // 8 MB
    const size_t SZ_W   = (size_t)MOE_E * MOE_I * MOE_H * 2;      // 8 MB each
    unsigned short* xbf    = (unsigned short*)(ws);
    unsigned short* w1t    = (unsigned short*)(ws + SZ_XBF);
    unsigned short* w2t    = (unsigned short*)(ws + SZ_XBF + SZ_W);
    char*           tail   = ws + SZ_XBF + 2 * SZ_W;
    int*   rowTok  = (int*)  (tail);
    float* rowWgt  = (float*)(tail + 1 * 65536);
    int*   slotExp = (int*)  (tail + 2 * 65536);
    float* slotW   = (float*)(tail + 3 * 65536);
    int*   counts  = (int*)  (tail + 4 * 65536);
    int*   offsets = (int*)  (tail + 4 * 65536 + 256);
    int*   cursors = (int*)  (tail + 4 * 65536 + 512);

    hipMemsetAsync(d_out, 0, (size_t)out_size * sizeof(float), stream);
    hipMemsetAsync(counts, 0, 3 * 256, stream);

    moe_prep_x  <<<(MOE_T * MOE_H) / 256, 256, 0, stream>>>(x, xbf);
    moe_prep_w1t<<<(MOE_E * MOE_I * MOE_H) / 256, 256, 0, stream>>>(w1, w1t);
    moe_prep_w2t<<<(MOE_E * MOE_I * MOE_H) / 256, 256, 0, stream>>>(w2, w2t);
    moe_router  <<<MOE_T / 256, 256, 0, stream>>>(x, gw, logits, slotExp, slotW, counts);
    moe_scan    <<<1, 32, 0, stream>>>(counts, offsets, cursors);
    moe_build   <<<MOE_T / 256, 256, 0, stream>>>(slotExp, slotW, offsets, cursors,
                                                  rowTok, rowWgt);
    moe_expert  <<<dim3(MOE_T / TM, MOE_E), 256, 0, stream>>>(
        xbf, w1t, w2t, b1, b2, rowTok, rowWgt, counts, offsets, out);
}
